// ScaledDotProductAttention_85822036508845
// MI455X (gfx1250) — matrix operand order repaired and checked by reading, not compiled, not run on hardware
//
#include <hip/hip_runtime.h>
#include <hip/hip_bf16.h>

// CDNA5 / gfx1250 causal attention, fp32 in/out, f16 WMMA compute (f32 accum).
// attn (1 GiB) dominates traffic -> non-temporal streaming stores; score strip
// lives in the 320KB WGP LDS; V is staged with async global->LDS DMA
// (ASYNCcnt, wave-private double buffer, no barriers in the PV loop).

typedef __attribute__((ext_vector_type(16))) _Float16 v16h;
typedef __attribute__((ext_vector_type(8)))  float    v8f;
typedef __attribute__((ext_vector_type(4)))  float    f32x4;

#define SEQ   2048
#define HD    64
#define QT    16      // query rows per workgroup
#define NW    4       // waves per workgroup
#define NT    128     // threads per workgroup

__device__ __forceinline__ void async_copy_b128(unsigned lds_off, const void* gptr) {
    // GLOBAL_LOAD_ASYNC_TO_LDS_B128 (GV mode): per-lane 16B global -> LDS, ASYNCcnt.
    asm volatile("global_load_async_to_lds_b128 %0, %1, off"
                 :: "v"(lds_off), "v"(gptr) : "memory");
}
__device__ __forceinline__ void wait_async_le4() {
    asm volatile("s_wait_asynccnt 0x4" ::: "memory");
}
__device__ __forceinline__ void wait_async_0() {
    asm volatile("s_wait_asynccnt 0x0" ::: "memory");
}

__launch_bounds__(NT)
__global__ void fa_fwd_attn_kernel(const float* __restrict__ Q,
                                   const float* __restrict__ K,
                                   const float* __restrict__ V,
                                   float* __restrict__ Out,
                                   float* __restrict__ Attn)
{
    __shared__ float sc[QT * SEQ];           // 128 KB score strip (f32)
    __shared__ float wv[NW][2][32 * 16];     // 16 KB: per-wave double-buffered V slice
    __shared__ float red[QT * 8];
    __shared__ float rowmax[QT];
    __shared__ float rowsum[QT];

    const int qb   = blockIdx.x;             // 0..127 query block
    const int bh   = blockIdx.y;             // 0..63  (b*H + h)
    const int q0   = qb * QT;
    const int tid  = threadIdx.x;
    const int lane = tid & 31;
    const int wave = tid >> 5;               // 0..3
    const int g    = lane >> 4;              // lane group 0/1
    const int ln   = lane & 15;

    const float* Qh = Q + (size_t)bh * SEQ * HD;
    const float* Kh = K + (size_t)bh * SEQ * HD;
    const float* Vh = V + (size_t)bh * SEQ * HD;

    // ---------------- A fragments for Q tile (16x32 f16, two d-chunks), pre-scaled by 1/8
    v16h aq0, aq1;
    {
        const float* qrow = Qh + (size_t)(q0 + ln) * HD;
        #pragma unroll
        for (int h = 0; h < 8; ++h) {
            // A layout: elems 0..7 -> K = 8g..8g+7 ; elems 8..15 -> K = 16+8g..23+8g
            aq0[h]     = (_Float16)(qrow[      8*g + h     ] * 0.125f);
            aq0[8 + h] = (_Float16)(qrow[      16 + 8*g + h] * 0.125f);
            aq1[h]     = (_Float16)(qrow[32 +  8*g + h     ] * 0.125f);
            aq1[8 + h] = (_Float16)(qrow[32 + 16 + 8*g + h ] * 0.125f);
        }
    }

    // ---------------- phase 1: scores = (Q/8) . K^T for tiles up to the causal diagonal
    for (int kt = wave; kt <= qb; kt += NW) {
        const float* krow = Kh + (size_t)(kt * 16 + ln) * HD + g * 16;
        __builtin_prefetch(krow + (size_t)NW * 16 * HD, 0, 3);   // next tile for this wave
        // B fragment: lanes = key column n, elems 0..7 -> d = 8g+h, 8..15 -> d = 16+8g+h (same order as A)
        v16h bk0, bk1;
        #pragma unroll
        for (int h = 0; h < 8; ++h) {
            bk0[h]     = (_Float16)krow[h - 8 * g];             // d-chunk 0:  d = 8g + h
            bk0[8 + h] = (_Float16)krow[16 - 8 * g + h];        //             d = 16 + 8g + h
            bk1[h]     = (_Float16)krow[32 + h - 8 * g];        // d-chunk 1:  d = 32 + 8g + h
            bk1[8 + h] = (_Float16)krow[32 + 16 - 8 * g + h];   //             d = 48 + 8g + h
        }
        v8f acc = {};
        acc = __builtin_amdgcn_wmma_f32_16x16x32_f16(false, aq0, false, bk0,
                                                     (short)0, acc, false, false);
        acc = __builtin_amdgcn_wmma_f32_16x16x32_f16(false, aq1, false, bk1,
                                                     (short)0, acc, false, false);
        // C layout: VGPR v, group g -> M = v + 8g, N = ln
        #pragma unroll
        for (int vv = 0; vv < 8; ++vv)
            sc[(vv + 8 * g) * SEQ + kt * 16 + ln] = acc[vv];
    }
    __syncthreads();

    // ---------------- phase 2: softmax statistics (8 threads per row)
    const int lane8 = tid & 7;
    const int r     = tid >> 3;              // 0..15
    const int qg    = q0 + r;                // causal horizon of this row
    const int nchunks = qb / 2 + 1;          // 32-wide P.V chunks needed
    const int kend    = nchunks * 32;

    float mloc = -INFINITY;
    for (int j = lane8; j <= qg; j += 8)
        mloc = fmaxf(mloc, sc[r * SEQ + j]);
    red[r * 8 + lane8] = mloc;
    __syncthreads();
    if (tid < QT) {
        float mm = red[tid * 8];
        #pragma unroll
        for (int i = 1; i < 8; ++i) mm = fmaxf(mm, red[tid * 8 + i]);
        rowmax[tid] = mm;
    }
    __syncthreads();

    const float mrow = rowmax[r];
    float lsum = 0.f;
    for (int j = lane8; j < kend; j += 8) {
        float e = 0.f;
        if (j <= qg) {
            e = __expf(sc[r * SEQ + j] - mrow);
            lsum += e;
        }
        sc[r * SEQ + j] = e;                 // unnormalized probs (zeros past diagonal)
    }
    red[r * 8 + lane8] = lsum;
    __syncthreads();
    if (tid < QT) {
        float ss = 0.f;
        #pragma unroll
        for (int i = 0; i < 8; ++i) ss += red[tid * 8 + i];
        rowsum[tid] = ss;
    }
    __syncthreads();

    // ---- kick off async DMA of V chunk 0 (this wave's 16-wide d slice) before
    //      the attn streaming stores, so the copy hides under them.
    {
        int j = lane;                        // 4 x b128 per lane = 2 KB slice
        #pragma unroll
        for (int i = 0; i < 4; ++i, j += 32) {
            const int row = j >> 2, seg = j & 3;
            const char* gp = (const char*)(Vh + (size_t)row * HD + wave * 16) + seg * 16;
            unsigned lo = (unsigned)(uintptr_t)&wv[wave][0][row * 16 + seg * 4];
            async_copy_b128(lo, gp);
        }
    }

    // ---------------- phase 3: stream normalized attn rows to global (non-temporal)
    for (int r2 = 0; r2 < QT; ++r2) {
        const int   qg2 = q0 + r2;
        const float inv = 1.0f / rowsum[r2];
        float* arow = Attn + ((size_t)bh * SEQ + (size_t)qg2) * SEQ;
        for (int j = tid * 4; j < SEQ; j += NT * 4) {
            f32x4 p;
            #pragma unroll
            for (int c = 0; c < 4; ++c)
                p[c] = (j + c <= qg2) ? sc[r2 * SEQ + j + c] * inv : 0.0f;
            __builtin_nontemporal_store(p, (f32x4*)(arow + j));
        }
    }

    // ---------------- phase 4: O = P . V  (wave w owns d-slice [16w, 16w+16))
    // Wave-private async double buffer: no workgroup barriers in this loop.
    v8f o = {};
    for (int c = 0; c < nchunks; ++c) {
        const int buf = c & 1;
        if (c + 1 < nchunks) {
            int j = lane;
            #pragma unroll
            for (int i = 0; i < 4; ++i, j += 32) {
                const int row = j >> 2, seg = j & 3;
                const char* gp = (const char*)(Vh + (size_t)((c + 1) * 32 + row) * HD
                                               + wave * 16) + seg * 16;
                unsigned lo = (unsigned)(uintptr_t)&wv[wave][buf ^ 1][row * 16 + seg * 4];
                async_copy_b128(lo, gp);
            }
            wait_async_le4();                // chunk c's 4 ops retired (in-order)
        } else {
            wait_async_0();
        }

        // A fragment from probability strip (f32 -> f16 on the fly)
        v16h ap;
        const float* prow = sc + (size_t)ln * SEQ + c * 32;
        #pragma unroll
        for (int h = 0; h < 8; ++h) {
            ap[h]     = (_Float16)prow[8 * g + h];
            ap[8 + h] = (_Float16)prow[16 + 8 * g + h];
        }
        // B fragment: lane n = d column (wave*16+ln), kpos 8g+h / 16+8g+h
        v16h bv;
        const float* vsl = &wv[wave][buf][0];
        #pragma unroll
        for (int h = 0; h < 8; ++h) {
            bv[h]     = (_Float16)vsl[(8 * g + h) * 16 + ln];
            bv[8 + h] = (_Float16)vsl[(16 + 8 * g + h) * 16 + ln];
        }

        o = __builtin_amdgcn_wmma_f32_16x16x32_f16(false, ap, false, bv,
                                                   (short)0, o, false, false);
    }

    // final: fold 1/l normalization into output store
    #pragma unroll
    for (int vv = 0; vv < 8; ++vv) {
        const int m = vv + 8 * g;
        Out[((size_t)bh * SEQ + (size_t)(q0 + m)) * HD + wave * 16 + ln] =
            o[vv] * (1.0f / rowsum[m]);
    }
}

extern "C" void kernel_launch(void* const* d_in, const int* in_sizes, int n_in,
                              void* d_out, int out_size, void* d_ws, size_t ws_size,
                              hipStream_t stream) {
    const float* q = (const float*)d_in[0];
    const float* k = (const float*)d_in[1];
    const float* v = (const float*)d_in[2];
    // d_in[3] is the causal mask; causal structure is applied analytically.
    float* out  = (float*)d_out;                                  // [4,16,2048,64]
    float* attn = out + (size_t)4 * 16 * 2048 * 64;               // [4,16,2048,2048]
    dim3 grid(SEQ / QT, 4 * 16);   // 128 query blocks x 64 heads
    fa_fwd_attn_kernel<<<grid, NT, 0, stream>>>(q, k, v, out, attn);
}
